// PositiveWaveFunction_76476187672844
// MI455X (gfx1250) — matrix operand order repaired and checked by reading, not compiled, not run on hardware
//
#include <hip/hip_runtime.h>

// ---------------------------------------------------------------------------
// PositiveWaveFunction RNN forward for MI455X (gfx1250, wave32, WMMA).
//
// Heavy op per step: Hnew = tanh(Hold @ Whh^T + bias_sel)  [8192 x 1024]
// -> bf16 WMMA 16x16x32, f32 accumulation. Whh (2MB bf16) + h ping-pong
// (2x16MB bf16) are L2-resident, so the step is compute bound; the kernel
// therefore maximizes back-to-back v_wmma issue:
//   * wave tile 32M x 64N  (8 accumulators, 8 WMMA per K-step of 32)
//   * B^T tile double-buffered in LDS, staged with the CDNA5 async copy
//     global_load_async_to_lds_b128 (ASYNCcnt) overlapping compute
//   * all 4 B fragments preloaded to registers before the WMMA burst
// ---------------------------------------------------------------------------

typedef __bf16 bf16;
typedef __attribute__((ext_vector_type(16))) __bf16 v16bf;
typedef __attribute__((ext_vector_type(8)))  float  v8f;

#define NSPIN 32
#define BATCH 8192
#define HID   1024

// LDS tile: 32 K-rows x (64+8) padded bf16 cols; row stride 144B (16B mult).
#define LDS_ROW   72
#define LDS_TILE  (32 * LDS_ROW)

union FragBF {
    v16bf v;
    uint4 q[2];
};

// Async 16B global -> LDS copy (per-lane), tracked by ASYNCcnt.
// Low 32 bits of a generic pointer into __shared__ are the wave-relative LDS
// offset (ISA 10.2: flat LDS addr = addr[31:0]), which is what VDST wants.
__device__ __forceinline__ void async_ld128(bf16* lds_dst, const bf16* gsrc) {
    unsigned loff = (unsigned)(size_t)lds_dst;
    asm volatile("global_load_async_to_lds_b128 %0, %1, off"
                 :: "v"(loff), "v"(gsrc) : "memory");
}

__device__ __forceinline__ void wait_async0() {
#if __has_builtin(__builtin_amdgcn_s_wait_asynccnt)
    __builtin_amdgcn_s_wait_asynccnt(0);
#else
    asm volatile("s_wait_asynccnt 0x0" ::: "memory");
#endif
}

// ---------------------------------------------------------------------------
// Prep 1: Whh^T in bf16 (B fragments become contiguous row reads), and the
// two one-hot-selected bias vectors v0/v1 = W_ih[:,j] + b_ih + b_hh.
// ---------------------------------------------------------------------------
__global__ __launch_bounds__(256)
void prep_weights_kernel(const float* __restrict__ Whh,
                         const float* __restrict__ Wih,
                         const float* __restrict__ bih,
                         const float* __restrict__ bhh,
                         bf16* __restrict__ WhhT,
                         float* __restrict__ v0,
                         float* __restrict__ v1)
{
    int idx = blockIdx.x * 256 + threadIdx.x;      // 0 .. 1024*1024-1
    int r = idx >> 10;                             // row n of Whh
    int c = idx & (HID - 1);                       // col k of Whh
    WhhT[(size_t)c * HID + r] = (bf16)Whh[idx];    // WhhT[k][n] = Whh[n][k]
    if (idx < HID) {
        float bb = bih[idx] + bhh[idx];
        v0[idx] = Wih[idx * 2 + 0] + bb;
        v1[idx] = Wih[idx * 2 + 1] + bb;
    }
}

// ---------------------------------------------------------------------------
// Prep 2: per-batch spin bits (bias selector) and factorial permutation
// weights w_up/w_dn per step (includes +EPS; step 0 stores 1.0).
// ---------------------------------------------------------------------------
__global__ __launch_bounds__(256)
void prep_batch_kernel(const float* __restrict__ data_in,   // [32][8192][2]
                       unsigned* __restrict__ bitsel,       // [32][8192]
                       float* __restrict__ wup,             // [32][8192]
                       float* __restrict__ wdn)             // [32][8192]
{
    int b = blockIdx.x * 256 + threadIdx.x;
    if (b >= BATCH) return;

    float fact[NSPIN + 1];
    fact[0] = 1.0f;
    #pragma unroll
    for (int i = 1; i <= NSPIN; ++i) fact[i] = fact[i - 1] * (float)i;

    int cu = 0, cd = 0;
    for (int s = 0; s < NSPIN; ++s) {
        int bit = 0;
        if (s != 0) {
            // shifted input: x at step s is data_in[s-1]; one-hot chan 1 == bit
            float d1 = data_in[((size_t)(s - 1) * BATCH + b) * 2 + 1];
            bit = (d1 > 0.5f) ? 1 : 0;
            cu += (bit == 0);
            cd += (bit == 1);
        }
        bitsel[(size_t)s * BATCH + b] = (unsigned)bit;

        float WU = 1.0f, WD = 1.0f;
        if (s != 0) {
            int n  = NSPIN - s;          // <= 31
            int ku = NSPIN / 2 - cu;
            int kd = NSPIN / 2 - cd;
            float pu = (ku >= 0) ? (fact[n] / fact[ku]) : 0.0f;
            float pd = (kd >= 0) ? (fact[n] / fact[kd]) : 0.0f;
            WU = pu + 1e-12f;
            WD = pd + 1e-12f;
        }
        wup[(size_t)s * BATCH + b] = WU;
        wdn[(size_t)s * BATCH + b] = WD;
    }
}

// ---------------------------------------------------------------------------
// Recurrent GEMM step: Hout = tanh(Hin @ Whh^T + v_{bit}) in bf16.
// Block = 8 waves, block tile 256M x 64N; wave tile 32M x 64N.
// ---------------------------------------------------------------------------
__global__ __launch_bounds__(256)
void rnn_step_gemm_kernel(const bf16* __restrict__ Hin,
                          const bf16* __restrict__ WhhT,
                          const float* __restrict__ v0,
                          const float* __restrict__ v1,
                          const unsigned* __restrict__ bitsel,
                          bf16* __restrict__ Hout,
                          int step)
{
    __shared__ bf16 ldsB[2 * LDS_TILE];            // double-buffered B^T tile

    const int tid  = threadIdx.x;
    const int wave = tid >> 5;
    const int lane = tid & 31;
    const int hl   = lane >> 4;                    // half-wave index
    const int l16  = lane & 15;

    const int m0 = blockIdx.x * 256 + wave * 32;   // wave owns rows m0..m0+31
    const int n0 = blockIdx.y * 64;

    // staging assignment: thread t copies 16B of WhhT row (k0+kr), cols cc..cc+7
    const int kr = tid >> 3;                       // 0..31
    const int cc = (tid & 7) * 8;                  // 0,8,..,56

    v8f acc[2][4];
    #pragma unroll
    for (int mi = 0; mi < 2; ++mi)
        #pragma unroll
        for (int j = 0; j < 4; ++j) acc[mi][j] = (v8f)0.0f;

    if (step > 0) {                                // step 0: h_prev == 0
        const int kIters = HID / 32;

        // prologue: stage tile 0 into buffer 0
        async_ld128(&ldsB[kr * LDS_ROW + cc],
                    WhhT + (size_t)kr * HID + n0 + cc);
        wait_async0();
        __syncthreads();

        for (int kk = 0; kk < kIters; ++kk) {
            const int k0 = kk * 32;
            bf16* cur = &ldsB[(kk & 1) * LDS_TILE];

            // stage next tile into the other buffer (overlaps compute)
            if (kk + 1 < kIters) {
                bf16* nxt = &ldsB[((kk + 1) & 1) * LDS_TILE];
                async_ld128(&nxt[kr * LDS_ROW + cc],
                            WhhT + (size_t)(k0 + 32 + kr) * HID + n0 + cc);
            }

            // A fragments (16-bit A 16x32 layout: two 16B chunks per lane)
            FragBF a[2];
            #pragma unroll
            for (int mi = 0; mi < 2; ++mi) {
                const bf16* ap =
                    Hin + (size_t)(m0 + mi * 16 + l16) * HID + k0 + hl * 8;
                a[mi].q[0] = *(const uint4*)(ap);
                a[mi].q[1] = *(const uint4*)(ap + 16);
            }

            // preload all B fragments (lane <-> N, regs <-> K) to registers
            FragBF bfr[4];
            #pragma unroll
            for (int j = 0; j < 4; ++j) {
                const bf16* bp = &cur[(hl * 8) * LDS_ROW + j * 16 + l16];
                #pragma unroll
                for (int s = 0; s < 8; ++s) {
                    bfr[j].v[s]     = bp[s * LDS_ROW];
                    bfr[j].v[8 + s] = bp[(16 + s) * LDS_ROW];
                }
            }

            // 8 back-to-back WMMAs
            #pragma unroll
            for (int mi = 0; mi < 2; ++mi)
                #pragma unroll
                for (int j = 0; j < 4; ++j)
                    acc[mi][j] = __builtin_amdgcn_wmma_f32_16x16x32_bf16(
                        false, a[mi].v, false, bfr[j].v,
                        (short)0, acc[mi][j], false, false);

            // make next tile's LDS writes visible to all waves
            wait_async0();
            __syncthreads();
        }
    }

    // Epilogue: bit-selected bias + tanh, store bf16 h for next step.
    #pragma unroll
    for (int mi = 0; mi < 2; ++mi) {
        #pragma unroll
        for (int j = 0; j < 4; ++j) {
            int n = n0 + j * 16 + l16;
            float bv0 = v0[n];
            float bv1 = v1[n];
            #pragma unroll
            for (int r = 0; r < 8; ++r) {
                int m = m0 + mi * 16 + hl * 8 + r;         // C/D layout row
                unsigned bit = bitsel[(size_t)step * BATCH + m];
                float val = acc[mi][j][r] + (bit ? bv1 : bv0);
                Hout[(size_t)m * HID + n] = (bf16)tanhf(val);
            }
        }
    }
}

// ---------------------------------------------------------------------------
// Head: 2 logits per row, softmax x permutation weights, normalize.
// One wave per batch row, shuffle reduction.
// ---------------------------------------------------------------------------
__global__ __launch_bounds__(256)
void rnn_step_head_kernel(const bf16* __restrict__ Hcur,
                          const float* __restrict__ Wlin,   // [2][1024]
                          const float* __restrict__ blin,   // [2]
                          const float* __restrict__ wup,
                          const float* __restrict__ wdn,
                          float* __restrict__ out,          // [32][8192][2]
                          int step)
{
    const int wave = threadIdx.x >> 5;
    const int lane = threadIdx.x & 31;
    const int row  = blockIdx.x * 8 + wave;

    const bf16* h = Hcur + (size_t)row * HID;
    float s0 = 0.0f, s1 = 0.0f;
    #pragma unroll 4
    for (int i = lane; i < HID; i += 32) {
        float hv = (float)h[i];
        s0 += hv * Wlin[i];
        s1 += hv * Wlin[HID + i];
    }
    #pragma unroll
    for (int off = 16; off > 0; off >>= 1) {
        s0 += __shfl_down(s0, off, 32);
        s1 += __shfl_down(s1, off, 32);
    }
    if (lane == 0) {
        float l0 = s0 + blin[0];
        float l1 = s1 + blin[1];
        float mx = fmaxf(l0, l1);
        // softmax normalization cancels against the final renormalization
        float p0 = __expf(l0 - mx) * wup[(size_t)step * BATCH + row];
        float p1 = __expf(l1 - mx) * wdn[(size_t)step * BATCH + row];
        float inv = 1.0f / (p0 + p1);
        size_t o = ((size_t)step * BATCH + row) * 2;
        out[o + 0] = p0 * inv;
        out[o + 1] = p1 * inv;
    }
}

// ---------------------------------------------------------------------------
// Host-side launcher (graph-capture safe: stream-ordered launches only).
// ---------------------------------------------------------------------------
extern "C" void kernel_launch(void* const* d_in, const int* in_sizes, int n_in,
                              void* d_out, int out_size, void* d_ws, size_t ws_size,
                              hipStream_t stream) {
    const float* data_in = (const float*)d_in[0];   // [32][8192][2]
    const float* W_ih    = (const float*)d_in[1];   // [1024][2]
    const float* W_hh    = (const float*)d_in[2];   // [1024][1024]
    const float* b_ih    = (const float*)d_in[3];   // [1024]
    const float* b_hh    = (const float*)d_in[4];   // [1024]
    const float* W_lin   = (const float*)d_in[5];   // [2][1024]
    const float* b_lin   = (const float*)d_in[6];   // [2]
    float* out = (float*)d_out;                     // [32][8192][2]

    char* ws = (char*)d_ws;
    size_t off = 0;
    auto carve = [&](size_t bytes) -> char* {
        char* p = ws + off;
        off = (off + bytes + 255) & ~(size_t)255;
        return p;
    };
    bf16*     WhhT   = (bf16*)    carve((size_t)HID * HID * sizeof(bf16));         // 2 MB
    float*    v0     = (float*)   carve(HID * sizeof(float));
    float*    v1     = (float*)   carve(HID * sizeof(float));
    unsigned* bitsel = (unsigned*)carve((size_t)NSPIN * BATCH * sizeof(unsigned)); // 1 MB
    float*    wup    = (float*)   carve((size_t)NSPIN * BATCH * sizeof(float));    // 1 MB
    float*    wdn    = (float*)   carve((size_t)NSPIN * BATCH * sizeof(float));    // 1 MB
    bf16*     HA     = (bf16*)    carve((size_t)BATCH * HID * sizeof(bf16));       // 16 MB
    bf16*     HB     = (bf16*)    carve((size_t)BATCH * HID * sizeof(bf16));       // 16 MB

    prep_weights_kernel<<<(HID * HID) / 256, 256, 0, stream>>>(
        W_hh, W_ih, b_ih, b_hh, WhhT, v0, v1);
    prep_batch_kernel<<<BATCH / 256, 256, 0, stream>>>(
        data_in, bitsel, wup, wdn);

    bf16* Hbuf[2] = {HA, HB};
    dim3 ggrid(BATCH / 256, HID / 64);   // 32 x 16 blocks
    for (int s = 0; s < NSPIN; ++s) {
        bf16* Hin  = Hbuf[s & 1];        // unread at s==0 (GEMM skipped)
        bf16* Hout = Hbuf[(s + 1) & 1];
        rnn_step_gemm_kernel<<<ggrid, 256, 0, stream>>>(
            Hin, WhhT, v0, v1, bitsel, Hout, s);
        rnn_step_head_kernel<<<BATCH / 8, 256, 0, stream>>>(
            Hout, W_lin, b_lin, wup, wdn, out, s);
    }
}
